// MultiHeadGraphAttentionLayer_86638080295672
// MI455X (gfx1250) — matrix operand order repaired and checked by reading, not compiled, not run on hardware
//
#include <hip/hip_runtime.h>

// Multi-head graph attention (B=8, N=2048, F=256, H=4, D=64), flash-attention
// style, bf16 WMMA (v_wmma_f32_16x16x32_bf16) with f32 softmax/accumulate.
// d_ws holds P = per-head projections in bf16: [B][H][N][D] = 8 MB.
// LDS tiles use padded strides (non-power-of-two) so every ds_load_b128
// fragment load and the transpose store are bank-conflict-free.

#define BATCH 8
#define SEQ   2048
#define FIN   256
#define HEADS 4
#define DHEAD 64
#define ALPHA 0.2f

#define LDK 72   // sK/sKT row stride (ushorts): 144B = 36 dwords, 9 coprime 16
#define LDP 68   // p-scratch row stride (floats): 272B, 4n mod 64 distinct
#define LDW 264  // sW row stride (ushorts): 528B = 132 dwords

typedef __attribute__((ext_vector_type(16))) __bf16 v16bf;
typedef __attribute__((ext_vector_type(8)))  float  v8f;

union ABu { v16bf v; unsigned int u[8]; };

// pack two f32 -> packed bf16 pair (low = lo, high = hi).
__device__ __forceinline__ unsigned int pack2(float lo, float hi) {
#if __has_builtin(__builtin_amdgcn_cvt_pk_bf16_f32)
    auto p = __builtin_amdgcn_cvt_pk_bf16_f32(lo, hi);
    return __builtin_bit_cast(unsigned int, p);
#else
    unsigned ul = __builtin_bit_cast(unsigned, lo) + 0x8000u;
    unsigned uh = __builtin_bit_cast(unsigned, hi) + 0x8000u;
    return __builtin_amdgcn_perm(uh, ul, 0x07060302u); // high halves of both
#endif
}
// single f32 -> bf16 bits (round-to-nearest, ties away)
__device__ __forceinline__ unsigned short bfb1(float f) {
    return (unsigned short)((__builtin_bit_cast(unsigned, f) + 0x8000u) >> 16);
}

__device__ __forceinline__ v8f wmma_bf16(v16bf a, v16bf b, v8f c) {
    return __builtin_amdgcn_wmma_f32_16x16x32_bf16(false, a, false, b,
                                                   (short)0, c, false, false);
}

// A-fragment (16x32 bf16, MxK) from row-major bf16 src (first row at src).
// Lane L<16: row L, K 0..7 & 16..23; lane L+16: row L, K 8..15 & 24..31.
__device__ __forceinline__ v16bf load_a_bf16(const unsigned short* src, int ld,
                                             int kbase, int lane) {
    const int row = lane & 15;
    const int kh  = (lane >> 4) << 3;
    ABu f;
#pragma unroll
    for (int v2 = 0; v2 < 8; ++v2) {
        int kk = kbase + ((v2 >> 2) << 4) + kh + ((v2 & 3) << 1);
        f.u[v2] = *(const unsigned int*)(src + (size_t)row * ld + kk);
    }
    return f.v;
}

// A-fragment built from row-major f32 source, converting pairs (1 perm each).
__device__ __forceinline__ v16bf load_a_f32(const float* src, int ld,
                                            int kbase, int lane) {
    const int row = lane & 15;
    const int kh  = (lane >> 4) << 3;
    ABu f;
#pragma unroll
    for (int v2 = 0; v2 < 8; ++v2) {
        int kk = kbase + ((v2 >> 2) << 4) + kh + ((v2 & 3) << 1);
        float2 x = *(const float2*)(src + (size_t)row * ld + kk);
        f.u[v2] = pack2(x.x, x.y);
    }
    return f.v;
}

// B-fragment (32x16 bf16, KxN) read from B-transposed row-major storage [N][K]:
// lane L<16: col L, K 0..7 & 16..23; lane L+16: col L, K 8..15 & 24..31 (pairs -> b32 loads).
__device__ __forceinline__ v16bf load_b(const unsigned short* srcBT, int ld,
                                        int nbase, int kbase, int lane) {
    const int n  = nbase + (lane & 15);
    const int kh = (lane >> 4) << 3;
    ABu f;
#pragma unroll
    for (int v2 = 0; v2 < 8; ++v2) {
        f.u[v2] = *(const unsigned int*)(srcBT + (size_t)n * ld + kbase + ((v2 >> 2) << 4) + kh + ((v2 & 3) << 1));
    }
    return f.v;
}

__device__ __forceinline__ v16bf ones_frag() {
    ABu f;
#pragma unroll
    for (int v2 = 0; v2 < 8; ++v2) f.u[v2] = 0x3F803F80u; // bf16 1.0 pair
    return f.v;
}

// ---------------- Kernel 1: P[b,h,n,d] = h[b,n,:] @ W[h,:,d]  (bf16 out) ----
// grid: (B*SEQ/64, HEADS), block: 128 (4 waves; wave w -> rows w*16..w*16+15)
__global__ __launch_bounds__(128) void gat_proj(const float* __restrict__ Hm,
                                                const float* __restrict__ Wm,
                                                unsigned short* __restrict__ P) {
    __shared__ unsigned short sW[DHEAD * LDW]; // W^T for this head: [d][k], bf16
    const int tid  = threadIdx.x;
    const int head = blockIdx.y;
    const float* Wh = Wm + (size_t)head * FIN * DHEAD;

    for (int i = tid; i < FIN * (DHEAD / 2); i += 128) {
        int k = i >> 5;            // 32 float2 per k-row
        int d = (i & 31) << 1;
        float2 w2 = *(const float2*)(Wh + (size_t)k * DHEAD + d);
        sW[(d + 0) * LDW + k] = bfb1(w2.x);
        sW[(d + 1) * LDW + k] = bfb1(w2.y);
    }
    __syncthreads();

    const int wv = tid >> 5, lane = tid & 31;
    const int ln = lane & 15, khalf = lane >> 4;
    const int grow = blockIdx.x * 64 + wv * 16;            // row in [0, B*SEQ)
    const float* hrow = Hm + (size_t)grow * FIN;

    v8f C[4];
#pragma unroll
    for (int t = 0; t < 4; ++t) C[t] = (v8f){};

    for (int k0 = 0; k0 < FIN; k0 += 32) {
        v16bf a = load_a_f32(hrow, FIN, k0, lane);
#pragma unroll
        for (int t = 0; t < 4; ++t)
            C[t] = wmma_bf16(a, load_b(sW, LDW, t * 16, k0, lane), C[t]);
    }

    const int bb = (blockIdx.x * 64) / SEQ;                // 64 | SEQ: no straddle
    const int nbase = grow - bb * SEQ + (khalf << 3);
    unsigned short* pout =
        P + (((size_t)bb * HEADS + head) * SEQ + nbase) * DHEAD + ln;
#pragma unroll
    for (int t = 0; t < 4; ++t)
#pragma unroll
        for (int r = 0; r < 8; ++r)
            pout[(size_t)r * DHEAD + t * 16] = bfb1(C[t][r]);
}

// ---------------- Kernel 2: flash attention over P, residual add ------------
// grid: (SEQ/64, BATCH*HEADS), block: 128. Wave w owns 16 query rows.
__global__ __launch_bounds__(128) void gat_attn(const unsigned short* __restrict__ P,
                                                const float* __restrict__ Hm,
                                                float* __restrict__ Out) {
    __shared__ unsigned short sK [64 * LDK];       // key tile, row-major [key][d]
    __shared__ unsigned short sKT[DHEAD * LDK];    // key tile, transposed [d][key]
    __shared__ float          sPf[4 * 16 * LDP];   // per-wave prob scratch, f32

    const int tid = threadIdx.x, lane = tid & 31, wv = tid >> 5;
    const int ln = lane & 15, khalf = lane >> 4;
    const int bh = blockIdx.y;                 // b*H + head
    const int bb = bh >> 2, head = bh & 3;
    const int q0 = blockIdx.x * 64 + wv * 16;
    const unsigned short* Pbh = P + (size_t)bh * SEQ * DHEAD;

    // persistent Q fragments (16 rows x 64 d)
    const v16bf aq0 = load_a_bf16(Pbh + (size_t)q0 * DHEAD, DHEAD, 0, lane);
    const v16bf aq1 = load_a_bf16(Pbh + (size_t)q0 * DHEAD, DHEAD, 32, lane);
    const v16bf ones = ones_frag();

    v8f O[4];
    float m[8], l[8];
#pragma unroll
    for (int t = 0; t < 4; ++t) O[t] = (v8f){};
#pragma unroll
    for (int r = 0; r < 8; ++r) { m[r] = -1e30f; l[r] = 0.f; }

    float* myP = sPf + wv * 16 * LDP;

    for (int j0 = 0; j0 < SEQ; j0 += 64) {
        __syncthreads(); // previous tile's readers done
        for (int i = tid; i < 64 * 32; i += 128) {  // 2048 dwords = 8 KB
            int key = i >> 5;
            int d   = (i & 31) << 1;
            unsigned u = *(const unsigned int*)(Pbh + (size_t)(j0 + key) * DHEAD + d);
            *(unsigned int*)(sK + key * LDK + d) = u;   // lanes: consecutive d
            sKT[(d + 0) * LDK + key] = (unsigned short)(u & 0xFFFFu); // 72-dword
            sKT[(d + 1) * LDK + key] = (unsigned short)(u >> 16);     // lane stride
        }
        if (j0 + 64 < SEQ && tid < 64)  // prefetch next tile (1 row = 1 line)
            __builtin_prefetch(Pbh + (size_t)(j0 + 64 + tid) * DHEAD, 0, 3);
        __syncthreads(); // tile ready

        // S = Q @ K^T  (16 x 64), then leakyrelu
        v8f S[4];
#pragma unroll
        for (int jt = 0; jt < 4; ++jt) {
            v8f acc = (v8f){};
            acc = wmma_bf16(aq0, load_b(sK, LDK, jt * 16, 0, lane), acc);
            acc = wmma_bf16(aq1, load_b(sK, LDK, jt * 16, 32, lane), acc);
            S[jt] = acc;
        }
#pragma unroll
        for (int jt = 0; jt < 4; ++jt)
#pragma unroll
            for (int r = 0; r < 8; ++r) {
                float s = S[jt][r];
                S[jt][r] = fmaxf(s, 0.f) + ALPHA * fminf(s, 0.f);
            }

        // running max: rows live in 16-lane half-groups of the C layout
        float sc[8];
#pragma unroll
        for (int r = 0; r < 8; ++r) {
            float v = fmaxf(fmaxf(S[0][r], S[1][r]), fmaxf(S[2][r], S[3][r]));
#pragma unroll
            for (int off = 1; off < 16; off <<= 1)
                v = fmaxf(v, __shfl_xor(v, off, 32));
            float mi = fmaxf(m[r], v);
            sc[r] = __expf(m[r] - mi);
            m[r]  = mi;
        }
#pragma unroll
        for (int jt = 0; jt < 4; ++jt)
#pragma unroll
            for (int r = 0; r < 8; ++r)
                S[jt][r] = __expf(S[jt][r] - m[r]);
#pragma unroll
        for (int t = 0; t < 4; ++t)
#pragma unroll
            for (int r = 0; r < 8; ++r) O[t][r] *= sc[r];

        // C-layout -> A-layout via per-wave f32 LDS scratch (no conversion VALU)
#pragma unroll
        for (int jt = 0; jt < 4; ++jt)
#pragma unroll
            for (int r = 0; r < 8; ++r)
                myP[((khalf << 3) + r) * LDP + jt * 16 + ln] = S[jt][r];
        __syncthreads();

        const v16bf ap0 = load_a_f32(myP, LDP, 0, lane);
        const v16bf ap1 = load_a_f32(myP, LDP, 32, lane);

        // row sums via p @ ones: D[m][n] == rowsum(m) for every n, i.e. exactly
        // the lane/VGPR slot where l[r] lives -> no shuffle reduction needed.
        v8f ls = (v8f){};
        ls = wmma_bf16(ap0, ones, ls);
        ls = wmma_bf16(ap1, ones, ls);
#pragma unroll
        for (int r = 0; r < 8; ++r) l[r] = l[r] * sc[r] + ls[r];

#pragma unroll
        for (int t = 0; t < 4; ++t) {
            O[t] = wmma_bf16(ap0, load_b(sKT, LDK, t * 16, 0, lane), O[t]);
            O[t] = wmma_bf16(ap1, load_b(sKT, LDK, t * 16, 32, lane), O[t]);
        }
    }

    // normalize, add residual h, store f32
    float inv[8];
#pragma unroll
    for (int r = 0; r < 8; ++r) inv[r] = __builtin_amdgcn_rcpf(l[r]);

    const int qb = q0 + (khalf << 3);
    const float* hb = Hm + ((size_t)bb * SEQ + qb) * FIN + head * DHEAD + ln;
    float* ob       = Out + ((size_t)bb * SEQ + qb) * FIN + head * DHEAD + ln;
#pragma unroll
    for (int t = 0; t < 4; ++t)
#pragma unroll
        for (int r = 0; r < 8; ++r) {
            size_t off = (size_t)r * FIN + t * 16;
            ob[off] = O[t][r] * inv[r] + hb[off];
        }
}

extern "C" void kernel_launch(void* const* d_in, const int* in_sizes, int n_in,
                              void* d_out, int out_size, void* d_ws, size_t ws_size,
                              hipStream_t stream) {
    const float* h   = (const float*)d_in[0];
    // d_in[1] = adj: dead input in the reference (drop_edge output unused) -> never read
    const float* W   = (const float*)d_in[2];
    float* out       = (float*)d_out;
    unsigned short* P = (unsigned short*)d_ws;   // bf16 [B][H][SEQ][DHEAD], 8 MB

    gat_proj<<<dim3((BATCH * SEQ) / 64, HEADS), 128, 0, stream>>>(h, W, P);
    gat_attn<<<dim3(SEQ / 64, BATCH * HEADS), 128, 0, stream>>>(P, h, out);
}
